// EgoRelGATLayer_455266533850
// MI455X (gfx1250) — matrix operand order repaired and checked by reading, not compiled, not run on hardware
//
#include <hip/hip_runtime.h>

// ---------------------------------------------------------------------------
// EgoRelGAT layer for MI455X (gfx1250, wave32, WMMA, TDM).
// Algebraic restructuring: q depends only on the ego row, so the two
// N x FIN x OUT GEMMs collapse into:
//   logits = Xn @ t  (N x 1024 x 16, bf16 WMMA)  + edge-type bias table z
//   S      = w_T @ Xn (128 x 8192 x 1024, bf16 WMMA), w_T[c*16+h, m] = alpha
// then tiny f32 epilogues (S@Wv per (c,h) block, sum over c, m@Wo).
// The S-GEMM double-buffers its B operand through LDS with the Tensor Data
// Mover (tensor_load_to_lds + s_wait_tensorcnt).
// ---------------------------------------------------------------------------

#define N_NODES 8192
#define FIN     1024
#define OUT_DIM 1024
#define NHEAD   16
#define HDIM    64
#define CT      8
#define EEMB    32
#define GEO     3
#define LN_EPS  1e-5f
#define NEG_INF -1e9f

typedef __attribute__((ext_vector_type(16))) __bf16 v16bf;
typedef __attribute__((ext_vector_type(8)))  __bf16 v8bf;
typedef __attribute__((ext_vector_type(8)))  float  v8f;
typedef __attribute__((ext_vector_type(4)))  unsigned int v4u;
typedef __attribute__((ext_vector_type(8)))  int v8i;
typedef __attribute__((ext_vector_type(4)))  int v4i;

#if defined(__has_builtin)
#if __has_builtin(__builtin_amdgcn_tensor_load_to_lds) && \
    __has_builtin(__builtin_amdgcn_s_wait_tensorcnt)
#define HAVE_TDM 1
#endif
#endif
#ifndef HAVE_TDM
#define HAVE_TDM 0
#endif

__device__ __forceinline__ unsigned short f2bf(float f) {
  unsigned int u = __builtin_bit_cast(unsigned int, f);
  unsigned int r = u + 0x7FFFu + ((u >> 16) & 1u);   // round-to-nearest-even
  return (unsigned short)(r >> 16);
}

__device__ __forceinline__ v16bf pack16(v8bf lo, v8bf hi) {
  v16bf r;
#pragma unroll
  for (int i = 0; i < 8; ++i) { r[i] = lo[i]; r[i + 8] = hi[i]; }
  return r;
}

// ---------------------------------------------------------------------------
// Kernel 1: LayerNorm; emit bf16 Xn in (a) row-major layout (WMMA A-operand
// loads) and (b) B-operand-swizzled layout (lane=col+16*((K>>3)&1), slot=(K&7)+8*(K>>4)),
// plus the f32 ego row.
// ---------------------------------------------------------------------------
__global__ void k_layernorm(const float* __restrict__ X,
                            const float* __restrict__ ln_g,
                            const float* __restrict__ ln_b,
                            unsigned short* __restrict__ xn_row,
                            unsigned short* __restrict__ xn_sw,
                            float* __restrict__ ego) {
  const int row = blockIdx.x;
  const int tid = threadIdx.x;                       // 256 threads, 4 elems each
  const float4 xv = ((const float4*)(X + (size_t)row * FIN))[tid];

  __shared__ float red[256];
  red[tid] = xv.x + xv.y + xv.z + xv.w;
  __syncthreads();
  for (int o = 128; o > 0; o >>= 1) { if (tid < o) red[tid] += red[tid + o]; __syncthreads(); }
  const float mu = red[0] * (1.0f / FIN);
  __syncthreads();

  const float dx = xv.x - mu, dy = xv.y - mu, dz = xv.z - mu, dw = xv.w - mu;
  red[tid] = dx * dx + dy * dy + dz * dz + dw * dw;
  __syncthreads();
  for (int o = 128; o > 0; o >>= 1) { if (tid < o) red[tid] += red[tid + o]; __syncthreads(); }
  const float rs = rsqrtf(red[0] * (1.0f / FIN) + LN_EPS);

  const float4 g4 = ((const float4*)ln_g)[tid];
  const float4 b4 = ((const float4*)ln_b)[tid];
  float y[4];
  y[0] = dx * rs * g4.x + b4.x;
  y[1] = dy * rs * g4.y + b4.y;
  y[2] = dz * rs * g4.z + b4.z;
  y[3] = dw * rs * g4.w + b4.w;

  const int kb = row >> 5, kin = row & 31;
  const int lane_hi = 16 * ((kin >> 3) & 1), kpack = (kin & 7) + 8 * (kin >> 4);
#pragma unroll
  for (int i = 0; i < 4; ++i) {
    const int f = tid * 4 + i;
    const unsigned short hb = f2bf(y[i]);
    xn_row[(size_t)row * FIN + f] = hb;
    const int ft = f >> 4, n = f & 15;
    xn_sw[(((size_t)(kb * 64 + ft)) * 32 + (n + lane_hi)) * 16 + kpack] = hb;
  }
  if (row == 0) {
    float4 e;
    e.x = y[0]; e.y = y[1]; e.z = y[2]; e.w = y[3];
    ((float4*)ego)[tid] = e;
  }
}

// ---------------------------------------------------------------------------
// Kernel 2: geo bias projection gbp = geo_bias @ geo_w   (N x 16)
// ---------------------------------------------------------------------------
__global__ void k_gbp(const float* __restrict__ geo_bias,
                      const float* __restrict__ geo_w,
                      float* __restrict__ gbp) {
  const int i = blockIdx.x * blockDim.x + threadIdx.x;   // N*16
  const int m = i >> 4, h = i & 15;
  float s = 0.f;
#pragma unroll
  for (int g = 0; g < GEO; ++g) s += geo_bias[m * GEO + g] * geo_w[g * NHEAD + h];
  gbp[i] = s;
}

// ---------------------------------------------------------------------------
// Kernel 3: zero the sparse attention-weight matrix w_T (bf16 128 x N)
// ---------------------------------------------------------------------------
__global__ void k_zero(unsigned int* __restrict__ p, int nwords) {
  const int i = blockIdx.x * blockDim.x + threadIdx.x;
  if (i < nwords) p[i] = 0u;
}

// ---------------------------------------------------------------------------
// Kernel 4 (single 1024-thread block):
//   valid/etype/rank (prefix scan), q = ego@Wq,
//   tT[h,f] = (Wk_h @ q_h)/8  (bf16, head-major for contiguous B loads),
//   z[c,h]  = (edge_tab @ (We_h @ q_h))/8
// ---------------------------------------------------------------------------
__global__ void k_prep(const float* __restrict__ A,
                       const int*   __restrict__ E,
                       const float* __restrict__ Wq,
                       const float* __restrict__ Wk,
                       const float* __restrict__ We,
                       const float* __restrict__ edge_tab,
                       const float* __restrict__ ego,
                       int* __restrict__ validv,
                       int* __restrict__ etype,
                       int* __restrict__ rankv,
                       unsigned short* __restrict__ tT,
                       float* __restrict__ z) {
  const int tid = threadIdx.x;                // 1024
  __shared__ int   ssum[1024];
  __shared__ float egos[1024];
  __shared__ float qsh[1024];
  __shared__ float ush[EEMB * NHEAD];

  // ---- valid / etype / rank (inclusive scan over 8192 flags) ----
  const int base = tid * 8;
  int loc[8];
  int csum = 0;
#pragma unroll
  for (int i = 0; i < 8; ++i) {
    const int m = base + i;
    const int ev = E[m];
    const int vld = (A[m] > 0.f) && (m != 0) && (ev >= 0);
    int e = ev < 0 ? 0 : (ev > (CT - 1) ? (CT - 1) : ev);
    validv[m] = vld;
    etype[m] = e;
    csum += vld;
    loc[i] = csum;
  }
  ssum[tid] = csum;
  __syncthreads();
  for (int off = 1; off < 1024; off <<= 1) {
    const int v = ssum[tid];
    const int vp = (tid >= off) ? ssum[tid - off] : 0;
    __syncthreads();
    ssum[tid] = v + vp;
    __syncthreads();
  }
  const int before = (tid > 0) ? ssum[tid - 1] : 0;
#pragma unroll
  for (int i = 0; i < 8; ++i) rankv[base + i] = before + loc[i] - 1;

  // ---- q = ego @ Wq ----
  egos[tid] = ego[tid];
  __syncthreads();
  float qacc = 0.f;
  for (int f = 0; f < FIN; ++f) qacc += egos[f] * Wq[(size_t)f * OUT_DIM + tid];
  qsh[tid] = qacc;
  __syncthreads();

  // ---- tT[h, f] = (1/8) * sum_d Wk[f, h*64+d] * q[h*64+d] ----
  for (int o = tid; o < NHEAD * FIN; o += 1024) {
    const int h = o >> 10, f = o & (FIN - 1);
    float s = 0.f;
#pragma unroll 8
    for (int d = 0; d < HDIM; ++d) s += Wk[(size_t)f * OUT_DIM + h * HDIM + d] * qsh[h * HDIM + d];
    tT[(size_t)h * FIN + f] = f2bf(s * 0.125f);
  }

  // ---- u[e, h] = We_h @ q_h ; z[c, h] = (edge_tab @ u)/8 ----
  if (tid < EEMB * NHEAD) {
    const int e = tid >> 4, h = tid & 15;
    float s = 0.f;
#pragma unroll 8
    for (int d = 0; d < HDIM; ++d) s += We[(size_t)e * OUT_DIM + h * HDIM + d] * qsh[h * HDIM + d];
    ush[tid] = s;
  }
  __syncthreads();
  if (tid < CT * NHEAD) {
    const int c = tid >> 4, h = tid & 15;
    float s = 0.f;
#pragma unroll
    for (int e = 0; e < EEMB; ++e) s += edge_tab[c * EEMB + e] * ush[e * NHEAD + h];
    z[tid] = s * 0.125f;
  }
}

// ---------------------------------------------------------------------------
// Kernel 5: logits = Xn @ t  via v_wmma_f32_16x16x32_bf16.
// One wave (32 threads) per 16-node tile; 16 heads = exactly the N dim.
// Epilogue adds edge-type bias z, rank-aligned geo bias, NEG mask.
// ---------------------------------------------------------------------------
__global__ void k_logits(const __bf16* __restrict__ xn_row,
                         const __bf16* __restrict__ tT,
                         const float* __restrict__ z,
                         const float* __restrict__ gbp,
                         const int* __restrict__ validv,
                         const int* __restrict__ etype,
                         const int* __restrict__ rankv,
                         float* __restrict__ logits) {
  const int lane = threadIdx.x;              // 32
  const int row0 = blockIdx.x * 16;
  const int n = lane & 15, half = lane >> 4;
  const __bf16* abase = xn_row + (size_t)(row0 + n) * FIN;  // A row = node (M = lane%16)
  const __bf16* bbase = tT + (size_t)n * FIN;               // B col = head (N = lane%16)

  v8f acc = {};
#pragma unroll 4
  for (int k0 = 0; k0 < FIN; k0 += 32) {
    __builtin_prefetch(abase + k0 + 128, 0, 0);
    const v16bf a = *(const v16bf*)(abase + k0 + half * 16);
    const v16bf b = *(const v16bf*)(bbase + k0 + half * 16);
    acc = __builtin_amdgcn_wmma_f32_16x16x32_bf16(false, a, false, b, (short)0, acc,
                                                  false, false);
  }

  // C layout: VGPR j holds (M = 8*half + j, N = lane&15)
#pragma unroll
  for (int j = 0; j < 8; ++j) {
    const int m = row0 + 8 * half + j;
    const int h = n;
    float lg = acc[j] + z[etype[m] * NHEAD + h];
    if (validv[m]) {
      int r = rankv[m];
      r = r < 0 ? 0 : (r > N_NODES - 1 ? N_NODES - 1 : r);
      lg += gbp[(size_t)r * NHEAD + h];
    } else {
      lg = NEG_INF;
    }
    logits[(size_t)m * NHEAD + h] = lg;
  }
}

// ---------------------------------------------------------------------------
// Kernel 6: softmax over nodes per head; scatter alpha into w_T[c*16+h, m]
// (bf16, row-major so WMMA A-loads are contiguous).
// ---------------------------------------------------------------------------
__global__ void k_softmax(const float* __restrict__ logits,
                          const int* __restrict__ validv,
                          const int* __restrict__ etype,
                          unsigned short* __restrict__ wT) {
  const int h = blockIdx.x, tid = threadIdx.x;   // 256
  __shared__ float red[256];

  float mx = -3.0e38f;
  for (int m = tid; m < N_NODES; m += 256) mx = fmaxf(mx, logits[(size_t)m * NHEAD + h]);
  red[tid] = mx;
  __syncthreads();
  for (int o = 128; o > 0; o >>= 1) { if (tid < o) red[tid] = fmaxf(red[tid], red[tid + o]); __syncthreads(); }
  const float M = red[0];
  __syncthreads();

  float s = 0.f;
  for (int m = tid; m < N_NODES; m += 256) s += __expf(logits[(size_t)m * NHEAD + h] - M);
  red[tid] = s;
  __syncthreads();
  for (int o = 128; o > 0; o >>= 1) { if (tid < o) red[tid] += red[tid + o]; __syncthreads(); }
  const float inv = 1.f / red[0];

  for (int m = tid; m < N_NODES; m += 256) {
    if (validv[m]) {
      const float a = __expf(logits[(size_t)m * NHEAD + h] - M) * inv;
      wT[(size_t)(etype[m] * NHEAD + h) * N_NODES + m] = f2bf(a);
    }
  }
}

// ---------------------------------------------------------------------------
// Kernel 7: S = w_T @ Xn   (128 x 8192 x 1024) via bf16 WMMA.
// A from row-major w_T (streamed global b128 loads).
// B double-buffered into LDS by the Tensor Data Mover:
//   for a fixed feature-tile nt, B k-tiles are 1KB rows spaced 64KB apart
//   -> 2D TDM descriptor: tile_dim0 = 128 x 8B, tile_dim1 = CHUNK,
//      tensor_dim0_stride = 8192 x 8B.  s_wait_tensorcnt(1) overlaps the
//      chunk c+1 DMA with WMMA on chunk c.
// ---------------------------------------------------------------------------
#define CHUNK 16                                    // k-blocks per TDM fetch

#if HAVE_TDM
__device__ __forceinline__ void tdm_issue_b(const __bf16* gptr, unsigned lds_off) {
  const unsigned long long ga = (unsigned long long)(const void*)gptr;
  v4u g0;
  g0[0] = 1u;                                       // count=1, user mode
  g0[1] = lds_off;                                  // LDS byte address
  g0[2] = (unsigned int)ga;                         // global_addr[31:0]
  g0[3] = (unsigned int)((ga >> 32) & 0x01FFFFFFu)  // global_addr[56:32]
        | (2u << 30);                               // type = 2 ("image")
  v8i g1;
  g1[0] = 3 << 16;                                  // data_size = 8 bytes
  g1[1] = 128 << 16;                                // tensor_dim0 = 128 units
  g1[2] = CHUNK << 16;                              // tensor_dim1 = CHUNK rows
  g1[3] = 128 << 16;                                // tile_dim0 = 128 units (1KB)
  g1[4] = CHUNK;                                    // tile_dim1 = CHUNK, tile_dim2 = 0
  g1[5] = 8192;                                     // tensor_dim0_stride = 64KB (8B units)
  g1[6] = 0;
  g1[7] = 0;
  const v4i gz4 = {0, 0, 0, 0};                     // groups 2/3 unused (2D tile)
  const v8i gz8 = {0, 0, 0, 0, 0, 0, 0, 0};
  __builtin_amdgcn_tensor_load_to_lds(g0, g1, gz4, gz4, gz8, 0);
}
#endif

__global__ void k_sgemm(const __bf16* __restrict__ wT,
                        const __bf16* __restrict__ xn_sw,
                        float* __restrict__ S) {
  const int lane = threadIdx.x;           // 32 (one wave; EXEC all ones for WMMA)
  const int tile = blockIdx.x;            // 8 ch-tiles x 64 feature-tiles
  const int mt = tile & 7, nt = tile >> 3;
  const int n = lane & 15, half = lane >> 4;
  const __bf16* abase = wT + (size_t)(mt * 16 + n) * N_NODES;
  v8f acc = {};

#if HAVE_TDM
  __shared__ __align__(16) __bf16 bbuf[2][CHUNK * 512];   // 2 x 16KB
  // B k-tile kb for feature-tile nt starts at xn_sw + (kb*64 + nt)*512 bf16.
  const __bf16* gB = xn_sw + (size_t)nt * 512;
  const unsigned lds0 = (unsigned)(unsigned long long)(&bbuf[0][0]);
  const unsigned lds1 = (unsigned)(unsigned long long)(&bbuf[1][0]);
  const int NCHUNK = (N_NODES / 32) / CHUNK;                // 16
  tdm_issue_b(gB, lds0);
  for (int c = 0; c < NCHUNK; ++c) {
    if (c + 1 < NCHUNK) {
      tdm_issue_b(gB + (size_t)(c + 1) * CHUNK * 32768, ((c + 1) & 1) ? lds1 : lds0);
      __builtin_amdgcn_s_wait_tensorcnt(1);                 // chunk c resident
    } else {
      __builtin_amdgcn_s_wait_tensorcnt(0);
    }
    const __bf16* bl = &bbuf[c & 1][0];
#pragma unroll 4
    for (int j = 0; j < CHUNK; ++j) {
      const int kb = c * CHUNK + j;
      __builtin_prefetch(abase + kb * 32 + 256, 0, 0);
      const v8bf alo = *(const v8bf*)(abase + kb * 32 + half * 8);
      const v8bf ahi = *(const v8bf*)(abase + kb * 32 + 16 + half * 8);
      const v16bf a = pack16(alo, ahi);
      const v16bf b = *(const v16bf*)(bl + j * 512 + lane * 16);
      acc = __builtin_amdgcn_wmma_f32_16x16x32_bf16(false, a, false, b, (short)0, acc,
                                                    false, false);
    }
  }
#else
#pragma unroll 2
  for (int kb = 0; kb < N_NODES / 32; ++kb) {
    __builtin_prefetch(abase + kb * 32 + 256, 0, 0);
    const v8bf alo = *(const v8bf*)(abase + kb * 32 + half * 8);
    const v8bf ahi = *(const v8bf*)(abase + kb * 32 + 16 + half * 8);
    const v16bf a = pack16(alo, ahi);
    const v16bf b = *(const v16bf*)(xn_sw + ((size_t)(kb * 64 + nt) * 32 + lane) * 16);
    acc = __builtin_amdgcn_wmma_f32_16x16x32_bf16(false, a, false, b, (short)0, acc,
                                                  false, false);
  }
#endif

#pragma unroll
  for (int j = 0; j < 8; ++j)
    S[(size_t)(mt * 16 + 8 * half + j) * FIN + nt * 16 + n] = acc[j];
}

// ---------------------------------------------------------------------------
// Kernel 8: c_type[c, h*64+d] = S[c*16+h, :] @ Wv[:, h*64+d]
// ---------------------------------------------------------------------------
__global__ void k_ctype(const float* __restrict__ S,
                        const float* __restrict__ Wv,
                        float* __restrict__ outC) {
  const int idx = blockIdx.x * blockDim.x + threadIdx.x;    // 8192
  const int c = idx >> 10, o = idx & 1023, h = o >> 6;
  const float* Sr = S + (size_t)(c * NHEAD + h) * FIN;
  float s = 0.f;
  for (int f = 0; f < FIN; ++f) s += Sr[f] * Wv[(size_t)f * OUT_DIM + o];
  outC[idx] = s;
}

// ---------------------------------------------------------------------------
// Kernel 9: m = sum_c c_type ; out = ego + m @ Wo
// ---------------------------------------------------------------------------
__global__ void k_out(const float* __restrict__ outC,
                      const float* __restrict__ ego,
                      const float* __restrict__ Wo,
                      float* __restrict__ out) {
  __shared__ float mv[OUT_DIM];
  const int tid = threadIdx.x;              // 256
  for (int i = tid; i < OUT_DIM; i += 256) {
    float s = 0.f;
#pragma unroll
    for (int c = 0; c < CT; ++c) s += outC[c * OUT_DIM + i];
    mv[i] = s;
  }
  __syncthreads();
  const int j = blockIdx.x * 256 + tid;     // grid = 4
  float s = ego[j];
  for (int i = 0; i < OUT_DIM; ++i) s += mv[i] * Wo[(size_t)i * OUT_DIM + j];
  out[j] = s;
}

// ---------------------------------------------------------------------------
// Workspace layout (bytes)
// ---------------------------------------------------------------------------
static constexpr size_t SZ_XN     = (size_t)N_NODES * FIN * 2;          // bf16
static constexpr size_t OFF_XNROW = 0;
static constexpr size_t OFF_XNSW  = OFF_XNROW + SZ_XN;
static constexpr size_t OFF_EGO   = OFF_XNSW + SZ_XN;                   // f32[1024]
static constexpr size_t OFF_TT    = OFF_EGO + 4096;                     // bf16[16*1024]
static constexpr size_t OFF_Z     = OFF_TT + (size_t)NHEAD * FIN * 2;   // f32[128]
static constexpr size_t OFF_GBP   = OFF_Z + 512;                        // f32[N*16]
static constexpr size_t OFF_VALID = OFF_GBP + (size_t)N_NODES * NHEAD * 4;
static constexpr size_t OFF_ETYPE = OFF_VALID + (size_t)N_NODES * 4;
static constexpr size_t OFF_RANK  = OFF_ETYPE + (size_t)N_NODES * 4;
static constexpr size_t OFF_LOG   = OFF_RANK + (size_t)N_NODES * 4;     // f32[N*16]
static constexpr size_t OFF_WT    = OFF_LOG + (size_t)N_NODES * NHEAD * 4;  // bf16[128*N]
static constexpr size_t OFF_S     = OFF_WT + (size_t)CT * NHEAD * N_NODES * 2; // f32[128*1024]
static constexpr size_t WS_TOTAL  = OFF_S + (size_t)CT * NHEAD * FIN * 4;

extern "C" void kernel_launch(void* const* d_in, const int* in_sizes, int n_in,
                              void* d_out, int out_size, void* d_ws, size_t ws_size,
                              hipStream_t stream) {
  (void)in_sizes; (void)n_in; (void)out_size;
  if (ws_size < WS_TOTAL) return;

  const float* X        = (const float*)d_in[0];
  const float* A        = (const float*)d_in[1];
  const int*   E        = (const int*)  d_in[2];
  const float* geo_bias = (const float*)d_in[3];
  const float* Wq       = (const float*)d_in[4];
  const float* Wk       = (const float*)d_in[5];
  const float* Wv       = (const float*)d_in[6];
  const float* Wo       = (const float*)d_in[7];
  const float* We       = (const float*)d_in[8];
  const float* edge_tab = (const float*)d_in[9];
  const float* geo_w    = (const float*)d_in[10];
  const float* ln_g     = (const float*)d_in[11];
  const float* ln_b     = (const float*)d_in[12];

  char* ws = (char*)d_ws;
  unsigned short* xn_row = (unsigned short*)(ws + OFF_XNROW);
  unsigned short* xn_sw  = (unsigned short*)(ws + OFF_XNSW);
  float*          ego    = (float*)(ws + OFF_EGO);
  unsigned short* tT     = (unsigned short*)(ws + OFF_TT);
  float*          z      = (float*)(ws + OFF_Z);
  float*          gbp    = (float*)(ws + OFF_GBP);
  int*            validv = (int*)(ws + OFF_VALID);
  int*            etype  = (int*)(ws + OFF_ETYPE);
  int*            rankv  = (int*)(ws + OFF_RANK);
  float*          logits = (float*)(ws + OFF_LOG);
  unsigned short* wT     = (unsigned short*)(ws + OFF_WT);
  float*          S      = (float*)(ws + OFF_S);
  float*          fout   = (float*)d_out;

  k_layernorm<<<N_NODES, 256, 0, stream>>>(X, ln_g, ln_b, xn_row, xn_sw, ego);
  k_gbp<<<(N_NODES * NHEAD) / 256, 256, 0, stream>>>(geo_bias, geo_w, gbp);
  {
    const int nwords = (int)((size_t)CT * NHEAD * N_NODES * 2 / 4);
    k_zero<<<(nwords + 255) / 256, 256, 0, stream>>>((unsigned int*)wT, nwords);
  }
  k_prep<<<1, 1024, 0, stream>>>(A, E, Wq, Wk, We, edge_tab, ego,
                                 validv, etype, rankv, tT, z);
  k_logits<<<N_NODES / 16, 32, 0, stream>>>((const __bf16*)xn_row, (const __bf16*)tT,
                                            z, gbp, validv, etype, rankv, logits);
  k_softmax<<<NHEAD, 256, 0, stream>>>(logits, validv, etype, wT);
  k_sgemm<<<512, 32, 0, stream>>>((const __bf16*)wT, (const __bf16*)xn_sw, S);
  k_ctype<<<(CT * OUT_DIM) / 256, 256, 0, stream>>>(S, Wv, fout + OUT_DIM);
  k_out<<<4, 256, 0, stream>>>(fout + OUT_DIM, ego, Wo, fout);
}
